// RecurrentEntityNetwork_78065325572453
// MI455X (gfx1250) — compile-verified, matrix operand order checked by reading, not run on hardware
//
#include <hip/hip_runtime.h>
#include <hip/hip_bf16.h>
#include <math.h>

typedef __attribute__((ext_vector_type(16))) _Float16 v16h;
typedef __attribute__((ext_vector_type(8)))  _Float16 v8h;
typedef __attribute__((ext_vector_type(8)))  float    v8f;

#define S_LEN 128
#define BSZ   1024
#define OBJD  37
#define QD    11
#define HID   256
#define NSLOT 20
#define NROWS (BSZ * NSLOT)   // 20480

// ---------------------------------------------------------------------------
// WMMA fragment helpers (gfx1250 wave32, 16-bit A/B layout per CDNA5 ISA 7.12.2)
//   lanes 0..15 : row = lane,    K-halves {k0+0..7, k0+16..23}
//   lanes 16..31: row = lane-16, K-halves {k0+8..15, k0+24..31}
// ---------------------------------------------------------------------------
__device__ __forceinline__ v8f zero8() {
  v8f z;
#pragma unroll
  for (int i = 0; i < 8; ++i) z[i] = 0.f;
  return z;
}

__device__ __forceinline__ v16h ld_frag16(const _Float16* base, int ld, int lane, int k0) {
  const int r   = lane & 15;
  const int kb0 = k0 + ((lane & 16) ? 8 : 0);
  const _Float16* p = base + r * ld;
  v8h lo = *reinterpret_cast<const v8h*>(p + kb0);
  v8h hi = *reinterpret_cast<const v8h*>(p + kb0 + 16);
  v16h out;
#pragma unroll
  for (int i = 0; i < 8; ++i) { out[i] = lo[i]; out[i + 8] = hi[i]; }
  return out;
}

// Issue async stage of one 32-col weight panel (32 rows x 256 k f16 = 16 KB)
// into LDS: 128 threads x 8 chunks x 16 B, tracked by ASYNCcnt (8 ops/wave).
__device__ __forceinline__ void stage_panel_issue(uint32_t dst_lds, const _Float16* wsrc, int tid) {
  const char* src = (const char*)wsrc;
#pragma unroll
  for (int i = 0; i < 8; ++i) {
    const int off = tid * 16 + i * 2048;
    asm volatile("global_load_async_to_lds_b128 %0, %1, off"
                 :: "v"(dst_lds + off), "v"(src + off) : "memory");
  }
}

// ---------------------------------------------------------------------------
// Fused 2-layer MLP GEMM kernel: Y = relu(X @ W1^T + b1) @ W2^T (raw, no bias).
// Block: 128 threads = 4 waves, 64 rows. Double-buffered 32-col weight panels
// async-staged in LDS and shared by all 4 waves; A fragments preloaded into
// registers once per stage; relu intermediate staged per-wave in LDS.
// ---------------------------------------------------------------------------
__global__ void mlp2_kernel(const _Float16* __restrict__ X,
                            const _Float16* __restrict__ W1, const float* __restrict__ b1,
                            const _Float16* __restrict__ W2, float* __restrict__ Y) {
  __shared__ __align__(16) _Float16 panel[2][32 * HID];   // 2 x 16 KB
  __shared__ __align__(16) _Float16 ldsT[4 * 16 * HID];   // 32 KB
  const int tid  = threadIdx.x;
  const int wave = tid >> 5, lane = tid & 31;
  const int nl   = lane & 15;
  const int mhi  = (lane & 16) ? 8 : 0;
  const int row0 = blockIdx.x * 64;
  const uint32_t plds0 = (uint32_t)(uintptr_t)&panel[0][0];
  const uint32_t plds1 = (uint32_t)(uintptr_t)&panel[1][0];

  const _Float16* Xw = X + (size_t)(row0 + wave * 16) * HID;
  _Float16*       Tw = ldsT + wave * 16 * HID;

  // Preload all A fragments (16 rows x 256 K) into registers (64 VGPRs).
  v16h afr[8];
#pragma unroll
  for (int k = 0; k < 8; ++k) afr[k] = ld_frag16(Xw, HID, lane, k * 32);

  // ---- Stage 1: T = relu(X @ W1^T + b1) ----
  stage_panel_issue(plds0, W1, tid);
  for (int p = 0; p < 8; ++p) {
    const _Float16* pb = &panel[p & 1][0];
    if (p < 7) {
      stage_panel_issue((p & 1) ? plds0 : plds1, W1 + (p + 1) * 32 * HID, tid);
      asm volatile("s_wait_asynccnt 0x8" ::: "memory");   // panel p landed
    } else {
      asm volatile("s_wait_asynccnt 0x0" ::: "memory");
    }
    __syncthreads();
    v8f acc[2];
#pragma unroll
    for (int t = 0; t < 2; ++t) acc[t] = zero8();
#pragma unroll
    for (int k = 0; k < 8; ++k) {
#pragma unroll
      for (int t = 0; t < 2; ++t) {
        v16h bf = ld_frag16(pb + 16 * t * HID, HID, lane, k * 32);
        acc[t] = __builtin_amdgcn_wmma_f32_16x16x32_f16(false, afr[k], false, bf,
                                                        (short)0, acc[t], false, false);
      }
    }
#pragma unroll
    for (int t = 0; t < 2; ++t) {
      const int n    = p * 32 + 16 * t + nl;
      const float bb = b1[n];
#pragma unroll
      for (int i = 0; i < 8; ++i) {
        float v = acc[t][i] + bb;
        Tw[(i + mhi) * HID + n] = (_Float16)(v > 0.f ? v : 0.f);
      }
    }
    __syncthreads();   // all waves done reading this panel before overwrite
  }

  // Preload relu-T fragments into registers (reuses A registers).
  v16h tfr[8];
#pragma unroll
  for (int k = 0; k < 8; ++k) tfr[k] = ld_frag16(Tw, HID, lane, k * 32);

  // ---- Stage 2: Y = T @ W2^T ----
  float* Yw = Y + (size_t)(row0 + wave * 16) * HID;
  stage_panel_issue(plds0, W2, tid);
  for (int p = 0; p < 8; ++p) {
    const _Float16* pb = &panel[p & 1][0];
    if (p < 7) {
      stage_panel_issue((p & 1) ? plds0 : plds1, W2 + (p + 1) * 32 * HID, tid);
      asm volatile("s_wait_asynccnt 0x8" ::: "memory");
    } else {
      asm volatile("s_wait_asynccnt 0x0" ::: "memory");
    }
    __syncthreads();
    v8f acc[2];
#pragma unroll
    for (int t = 0; t < 2; ++t) acc[t] = zero8();
#pragma unroll
    for (int k = 0; k < 8; ++k) {
#pragma unroll
      for (int t = 0; t < 2; ++t) {
        v16h bf = ld_frag16(pb + 16 * t * HID, HID, lane, k * 32);
        acc[t] = __builtin_amdgcn_wmma_f32_16x16x32_f16(false, tfr[k], false, bf,
                                                        (short)0, acc[t], false, false);
      }
    }
#pragma unroll
    for (int t = 0; t < 2; ++t) {
      const int n = p * 32 + 16 * t + nl;
#pragma unroll
      for (int i = 0; i < 8; ++i)
        Yw[(i + mhi) * HID + n] = acc[t][i];
    }
    __syncthreads();
  }
}

// ---------------------------------------------------------------------------
// Per-row tail: gate / PReLU / update / L2-normalize. 8 lanes per row,
// __shfl_xor reductions, refreshes f32 + f16 state. 32 rows / 256-thread block.
// ---------------------------------------------------------------------------
__global__ void row_update_kernel(float* __restrict__ jbuf, const float* __restrict__ J_b,
                                  const float* __restrict__ sent, const float* __restrict__ vcontrib,
                                  const float* __restrict__ kcontrib, const float* __restrict__ embed,
                                  const float* __restrict__ prelu,
                                  float* __restrict__ mem, _Float16* __restrict__ mem_h) {
  const int tid = threadIdx.x;
  const int r   = blockIdx.x * 32 + (tid >> 3);
  const int c0  = (tid & 7) * 32;
  const int b   = r / NSLOT;
  const int m   = r - b * NSLOT;
  const float pa = prelu[0];
  const float* sentb = sent + b * HID;
  const float* vb    = vcontrib + b * HID;
  const float* kb    = kcontrib + m * HID;
  const float* em    = embed + m * HID;
  float*    memr  = mem + (size_t)r * HID;
  _Float16* memhr = mem_h + (size_t)r * HID;
  float*    jr    = jbuf + (size_t)r * HID;

  float g = 0.f;
#pragma unroll 8
  for (int i = 0; i < 32; ++i) {
    const int c = c0 + i;
    g += sentb[c] * (memr[c] + em[c]);
  }
  g += __shfl_xor(g, 1); g += __shfl_xor(g, 2); g += __shfl_xor(g, 4);
  const float gate = 1.f / (1.f + __expf(-g));

  float u[32];
  float nrm = 0.f;
#pragma unroll 8
  for (int i = 0; i < 32; ++i) {
    const int c  = c0 + i;
    const float j  = jr[c] + J_b[c] + vb[c] + kb[c];
    const float cd = j > 0.f ? j : pa * j;            // PReLU
    const float uu = memr[c] + gate * cd;
    u[i] = uu;
    nrm += uu * uu;
  }
  nrm += __shfl_xor(nrm, 1); nrm += __shfl_xor(nrm, 2); nrm += __shfl_xor(nrm, 4);
  const float inv = 1.f / (sqrtf(nrm) + 1e-12f);
#pragma unroll 8
  for (int i = 0; i < 32; ++i) {
    const int c = c0 + i;
    const float val = u[i] * inv;
    memr[c]  = val;
    memhr[c] = (_Float16)val;
  }
}

// ---------------------------------------------------------------------------
// Prep kernels
// ---------------------------------------------------------------------------
__global__ void prep_weights_kernel(const float* __restrict__ U_w, const float* __restrict__ V_w,
                                    const float* __restrict__ J_w,
                                    _Float16* __restrict__ U_h, _Float16* __restrict__ V_h,
                                    _Float16* __restrict__ Ju_h, _Float16* __restrict__ Jv_h) {
  const int i = blockIdx.x * blockDim.x + threadIdx.x;   // 0..65535
  const int n = i >> 8, k = i & 255;
  U_h[i]  = (_Float16)U_w[i];
  V_h[i]  = (_Float16)V_w[i];
  Ju_h[i] = (_Float16)J_w[n * 768 + k];
  Jv_h[i] = (_Float16)J_w[n * 768 + 256 + k];
}

__global__ void q_kernel(const float* __restrict__ qin, const float* __restrict__ Q_w,
                         const float* __restrict__ Q_b, float* __restrict__ q) {
  const int b = blockIdx.x, d = threadIdx.x;
  const float* x = qin + b * QD;
  const float* w = Q_w + d * QD;
  float acc = Q_b[d];
#pragma unroll
  for (int i = 0; i < QD; ++i) acc += x[i] * w[i];
  q[b * HID + d] = acc > 0.f ? acc : 0.f;
}

// kcontrib[m][d] = sum_k relu(embed[m] @ W_w^T + W_b)[k] * J_w[d][512+k]
__global__ void kcontrib_kernel(const float* __restrict__ embed, const float* __restrict__ W_w,
                                const float* __restrict__ W_b, const float* __restrict__ J_w,
                                float* __restrict__ kcontrib) {
  __shared__ float kpart[HID];
  const int m = blockIdx.x, d = threadIdx.x;
  const float* e = embed + m * HID;
  const float* w = W_w + d * HID;
  float acc = W_b[d];
  for (int i = 0; i < HID; ++i) acc += e[i] * w[i];
  kpart[d] = acc > 0.f ? acc : 0.f;
  __syncthreads();
  const float* jrw = J_w + d * 768 + 512;
  float acc2 = 0.f;
  for (int k = 0; k < HID; ++k) acc2 += kpart[k] * jrw[k];
  kcontrib[m * HID + d] = acc2;
}

__global__ void init_mem_kernel(const float* __restrict__ embed,
                                float* __restrict__ mem, _Float16* __restrict__ mem_h) {
  const int idx = blockIdx.x * blockDim.x + threadIdx.x;
  const int r = idx >> 8, d = idx & 255;
  const int m = r % NSLOT;
  const float v = embed[m * HID + d];
  mem[idx]   = v;
  mem_h[idx] = (_Float16)v;
}

__global__ void sent_kernel(const float* __restrict__ minp_s, const float* __restrict__ C_w,
                            const float* __restrict__ C_b, float* __restrict__ sent,
                            _Float16* __restrict__ sent_h) {
  const int b = blockIdx.x, d = threadIdx.x;
  const float* x = minp_s + b * OBJD;
  const float* w = C_w + d * OBJD;
  float acc = C_b[d];
#pragma unroll
  for (int i = 0; i < OBJD; ++i) acc += x[i] * w[i];
  acc = acc > 0.f ? acc : 0.f;
  sent[b * HID + d]   = acc;
  sent_h[b * HID + d] = (_Float16)acc;
}

// ---------------------------------------------------------------------------
// Final attention + output head (tiny, f32 VALU)
// ---------------------------------------------------------------------------
__global__ void final_kernel(const float* __restrict__ mem, const float* __restrict__ q,
                             const float* __restrict__ H_w, const float* __restrict__ H_b,
                             float* __restrict__ out) {
  __shared__ float e[NSLOT];
  __shared__ float attn[NSLOT];
  __shared__ float att[HID];
  const int b = blockIdx.x, d = threadIdx.x;
  const float* qr = q + b * HID;

  if (d < NSLOT) {
    const float* mr = mem + (size_t)(b * NSLOT + d) * HID;
    float acc = 0.f;
    for (int c = 0; c < HID; ++c) acc += mr[c] * qr[c];
    e[d] = acc;
  }
  __syncthreads();
  if (d == 0) {
    float mx = e[0];
    for (int m = 1; m < NSLOT; ++m) mx = e[m] > mx ? e[m] : mx;
    float s = 0.f;
    for (int m = 0; m < NSLOT; ++m) { attn[m] = __expf(e[m] - mx); s += attn[m]; }
    const float inv = 1.f / s;
    for (int m = 0; m < NSLOT; ++m) attn[m] *= inv;
  }
  __syncthreads();
  float a = 0.f;
  for (int m = 0; m < NSLOT; ++m) a += mem[(size_t)(b * NSLOT + m) * HID + d] * attn[m];
  att[d] = a;
  __syncthreads();
  const float* hw = H_w + d * (2 * HID);
  float acc = H_b[d];
  for (int k = 0; k < HID; ++k) acc += qr[k] * hw[k];
  for (int k = 0; k < HID; ++k) acc += att[k] * hw[HID + k];
  out[b * HID + d] = acc > 0.f ? acc : 0.f;
}

// ---------------------------------------------------------------------------
// Host launch
// ---------------------------------------------------------------------------
extern "C" void kernel_launch(void* const* d_in, const int* in_sizes, int n_in,
                              void* d_out, int out_size, void* d_ws, size_t ws_size,
                              hipStream_t stream) {
  (void)in_sizes; (void)n_in; (void)out_size; (void)ws_size;
  const float* minp  = (const float*)d_in[0];
  const float* qin   = (const float*)d_in[1];
  const float* C_w   = (const float*)d_in[2];
  const float* C_b   = (const float*)d_in[3];
  const float* Q_w   = (const float*)d_in[4];
  const float* Q_b   = (const float*)d_in[5];
  const float* H_w   = (const float*)d_in[6];
  const float* H_b   = (const float*)d_in[7];
  const float* U_w   = (const float*)d_in[8];
  const float* U_b   = (const float*)d_in[9];
  const float* V_w   = (const float*)d_in[10];
  const float* V_b   = (const float*)d_in[11];
  const float* W_w   = (const float*)d_in[12];
  const float* W_b   = (const float*)d_in[13];
  const float* J_w   = (const float*)d_in[14];
  const float* J_b   = (const float*)d_in[15];
  const float* prelu = (const float*)d_in[16];
  const float* embed = (const float*)d_in[17];
  float* out = (float*)d_out;

  char* ws = (char*)d_ws;
  size_t off = 0;
  auto take = [&](size_t bytes) -> void* {
    void* p = ws + off;
    off = (off + bytes + 255) & ~(size_t)255;
    return p;
  };
  float*    mem      = (float*)take((size_t)NROWS * HID * 4);
  _Float16* mem_h    = (_Float16*)take((size_t)NROWS * HID * 2);
  float*    jbuf     = (float*)take((size_t)NROWS * HID * 4);
  float*    sent     = (float*)take((size_t)BSZ * HID * 4);
  _Float16* sent_h   = (_Float16*)take((size_t)BSZ * HID * 2);
  float*    vcontrib = (float*)take((size_t)BSZ * HID * 4);
  float*    qbuf     = (float*)take((size_t)BSZ * HID * 4);
  float*    kcontrib = (float*)take((size_t)NSLOT * HID * 4);
  _Float16* U_h      = (_Float16*)take((size_t)HID * HID * 2);
  _Float16* V_h      = (_Float16*)take((size_t)HID * HID * 2);
  _Float16* Ju_h     = (_Float16*)take((size_t)HID * HID * 2);
  _Float16* Jv_h     = (_Float16*)take((size_t)HID * HID * 2);

  prep_weights_kernel<<<(HID * HID) / 256, 256, 0, stream>>>(U_w, V_w, J_w, U_h, V_h, Ju_h, Jv_h);
  q_kernel<<<BSZ, HID, 0, stream>>>(qin, Q_w, Q_b, qbuf);
  kcontrib_kernel<<<NSLOT, HID, 0, stream>>>(embed, W_w, W_b, J_w, kcontrib);
  init_mem_kernel<<<(NROWS * HID) / 256, 256, 0, stream>>>(embed, mem, mem_h);

  for (int s = 0; s < S_LEN; ++s) {
    sent_kernel<<<BSZ, HID, 0, stream>>>(minp + (size_t)s * BSZ * OBJD, C_w, C_b, sent, sent_h);
    mlp2_kernel<<<BSZ / 64, 128, 0, stream>>>(sent_h, V_h, V_b, Jv_h, vcontrib);
    mlp2_kernel<<<NROWS / 64, 128, 0, stream>>>(mem_h, U_h, U_b, Ju_h, jbuf);
    row_update_kernel<<<NROWS / 32, 256, 0, stream>>>(jbuf, J_b, sent, vcontrib, kcontrib,
                                                      embed, prelu, mem, mem_h);
  }

  final_kernel<<<BSZ, HID, 0, stream>>>(mem, qbuf, H_w, H_b, out);
}
